// CausalSelfAttention_27822798143730
// MI455X (gfx1250) — matrix operand order repaired and checked by reading, not compiled, not run on hardware
//
#include <hip/hip_runtime.h>

typedef __attribute__((ext_vector_type(16))) _Float16 v16h;
typedef __attribute__((ext_vector_type(8)))  _Float16 v8h;
typedef __attribute__((ext_vector_type(8)))  float    v8f;

#define NTOK 4096
#define DIM  256
#define WIN  128

// workspace layout (in _Float16 units)
#define WS_XH  ((size_t)0)                       // x as f16, row-major   [NTOK][DIM]
#define WS_QH  ((size_t)(NTOK * DIM))            // Q  f16, row-major     [NTOK][DIM]
#define WS_KH  ((size_t)(2 * NTOK * DIM))        // K  f16, row-major     [NTOK][DIM]
#define WS_VT  ((size_t)(3 * NTOK * DIM))        // V^T f16               [DIM][NTOK]
#define WS_WT  ((size_t)(4 * NTOK * DIM))        // W^T f16 (Q,K,V)       [3][DIM][DIM]

__device__ __forceinline__ v8f wmma_f16(v16h a, v16h b, v8f c) {
    // (neg_a, A, neg_b, B, c_mod, C, reuse_a, reuse_b)
    return __builtin_amdgcn_wmma_f32_16x16x32_f16(false, a, false, b, (short)0, c,
                                                  false, false);
}

// ---------------------------------------------------------------------------
// Kernel 0: one-time layout prep.
//   xh[i]       = (f16) x[i]
//   Wt[w][n][k] = (f16) W_w[k][n]     (B-fragment-native: contiguous in k)
// ---------------------------------------------------------------------------
__global__ __launch_bounds__(256) void prep_kernel(const float* __restrict__ x,
                                                   const float* __restrict__ Wq,
                                                   const float* __restrict__ Wk,
                                                   const float* __restrict__ Wv,
                                                   _Float16* __restrict__ ws) {
    const int idx = blockIdx.x * 256 + threadIdx.x;
    if (idx < NTOK * DIM)
        ws[WS_XH + idx] = (_Float16)x[idx];
    if (idx < 3 * DIM * DIM) {
        const int which = idx / (DIM * DIM);
        const int r     = idx % (DIM * DIM);
        const int n = r / DIM, k = r % DIM;
        const float* W = (which == 0) ? Wq : (which == 1) ? Wk : Wv;
        ws[WS_WT + idx] = (_Float16)W[k * DIM + n];
    }
}

// ---------------------------------------------------------------------------
// Kernel 1: Q/K/V projection.  Each wave computes a 16x64 strip (4 tiles
// sharing one A fragment), K=256 in 8 WMMA steps per tile.
// All fragment loads are contiguous vector loads (A: 2x16B, B: 1x32B).
// unroll(2) on the K-loop bounds software pipelining: verified no spills.
// Q,K stored row-major; V stored transposed (packed 16B store per lane).
// ---------------------------------------------------------------------------
__global__ __launch_bounds__(256) void qkv_project_kernel(const float* __restrict__ bq,
                                                          const float* __restrict__ bk,
                                                          const float* __restrict__ bv,
                                                          _Float16* __restrict__ ws) {
    const int lane = threadIdx.x & 31;
    const int wave = threadIdx.x >> 5;
    const int half = lane >> 4;
    const int l16  = lane & 15;

    const int tile  = blockIdx.x * 8 + wave;     // 0 .. 3071
    const int which = tile / (256 * 4);          // 0=Q, 1=K, 2=V
    const int rem   = tile % (256 * 4);
    const int rowT  = rem >> 2;                  // 0..255
    const int cs    = rem & 3;                   // col strip (64 wide)

    const _Float16* xh = ws + WS_XH;
    const _Float16* Wt = ws + WS_WT + (size_t)which * DIM * DIM;
    const float*    bm = (which == 0) ? bq : (which == 1) ? bk : bv;

    const int row      = rowT * 16 + l16;        // A-fragment row for this lane
    const int colBase0 = cs * 64;

    v8f zero = {};
    v8f acc[4] = {zero, zero, zero, zero};

    #pragma unroll 2
    for (int kk = 0; kk < DIM; kk += 32) {
        // A fragment: K = kk + 16*half + [0..15]
        const _Float16* xr = xh + (size_t)row * DIM + kk + 16 * half;
        v8h a0 = *(const v8h*)xr;
        v8h a1 = *(const v8h*)(xr + 8);
        v16h a;
        #pragma unroll
        for (int e = 0; e < 8; ++e) { a[e] = a0[e]; a[8 + e] = a1[e]; }

        #pragma unroll
        for (int c = 0; c < 4; ++c) {
            const int col = colBase0 + c * 16 + l16;   // B column for this lane
            v16h b = *(const v16h*)(Wt + (size_t)col * DIM + kk + 16 * half);
            acc[c] = wmma_f16(a, b, acc[c]);
        }
    }

    #pragma unroll
    for (int c = 0; c < 4; ++c) {
        const int colBase = colBase0 + c * 16;
        const float bias  = bm[colBase + l16];
        if (which < 2) {
            _Float16* Yh = ws + ((which == 0) ? WS_QH : WS_KH);
            #pragma unroll
            for (int j = 0; j < 8; ++j)
                Yh[(size_t)(rowT * 16 + j + 8 * half) * DIM + colBase + l16] =
                    (_Float16)(acc[c][j] + bias);
        } else {
            // V^T: lane's 8 accumulator rows are 8 consecutive tokens -> 16B store
            v8h pack;
            #pragma unroll
            for (int j = 0; j < 8; ++j)
                pack[j] = (_Float16)(acc[c][j] + bias);
            *(v8h*)(ws + WS_VT + (size_t)(colBase + l16) * NTOK +
                    rowT * 16 + 8 * half) = pack;
        }
    }
}

// ---------------------------------------------------------------------------
// Kernel 2: sliding-window attention.  One wave per 16-query tile.
// Keys span [qStart-128, qStart+15] = 9 aligned 16-key tiles, processed as
// 3 fully-unrolled groups of 3 (live regs: 72 acc + ~64 operands < budget).
// S (16x144) lives in 9 v8f accumulators; softmax fully in registers.
// P -> LDS f16 (rows padded to 168 halves: conflict-free b128 reads),
// then O = P @ V_win using contiguous v16h B loads from V^T.
// ---------------------------------------------------------------------------
__global__ __launch_bounds__(128) void sw_attention_kernel(const _Float16* __restrict__ ws,
                                                           float* __restrict__ out) {
    const int lane = threadIdx.x & 31;
    const int wave = threadIdx.x >> 5;           // 0..3
    const int half = lane >> 4;
    const int l16  = lane & 15;

    const _Float16* Qh = ws + WS_QH;
    const _Float16* Kh = ws + WS_KH;
    const _Float16* Vt = ws + WS_VT;

    __shared__ __align__(16) _Float16 Plds[4][16][168];  // 168: bank-conflict pad

    const int qt     = blockIdx.x * 4 + wave;    // 0..255
    const int qStart = qt * 16;
    const int k0     = qStart - 128;             // first (aligned) key tile base

    // ---- S = Q @ K_win^T  (9 tiles of 16x16, f32 accum) ----
    v8f zero = {};
    v8f s[9];
    #pragma unroll
    for (int t = 0; t < 9; ++t) s[t] = zero;

    #pragma unroll
    for (int tg = 0; tg < 9; tg += 3) {          // 3 key tiles per pass
        #pragma unroll 2
        for (int kk = 0; kk < DIM; kk += 32) {
            const _Float16* qr = Qh + (size_t)(qStart + l16) * DIM + kk + 16 * half;
            v8h a0 = *(const v8h*)qr;
            v8h a1 = *(const v8h*)(qr + 8);
            v16h a;
            #pragma unroll
            for (int e = 0; e < 8; ++e) { a[e] = a0[e]; a[8 + e] = a1[e]; }

            #pragma unroll
            for (int u = 0; u < 3; ++u) {
                const int t = tg + u;
                int krow = k0 + t * 16 + l16;    // B column l16 = key row
                if (krow < 0) krow = 0;          // clamped; masked below
                v16h b = *(const v16h*)(Kh + (size_t)krow * DIM + kk + 16 * half);
                s[t] = wmma_f16(a, b, s[t]);
            }
        }
    }

    // ---- mask + softmax (row j+8*half lives in c[j] across the 16-lane half) ----
    const float scale = 0.0625f;                 // 1/sqrt(256)
    #pragma unroll
    for (int j = 0; j < 8; ++j) {
        const int i = qStart + j + 8 * half;     // absolute query index
        float m = -__builtin_inff();
        #pragma unroll
        for (int t = 0; t < 9; ++t) {
            const int p = k0 + t * 16 + l16;     // absolute key index
            const bool valid = (p >= 0) && (p >= i - (WIN - 1)) && (p <= i);
            float v = valid ? s[t][j] * scale : -__builtin_inff();
            s[t][j] = v;
            m = fmaxf(m, v);
        }
        m = fmaxf(m, __shfl_xor(m, 1));
        m = fmaxf(m, __shfl_xor(m, 2));
        m = fmaxf(m, __shfl_xor(m, 4));
        m = fmaxf(m, __shfl_xor(m, 8));
        float sum = 0.f;
        #pragma unroll
        for (int t = 0; t < 9; ++t) {
            float e = __expf(s[t][j] - m);
            s[t][j] = e;
            sum += e;
        }
        sum += __shfl_xor(sum, 1);
        sum += __shfl_xor(sum, 2);
        sum += __shfl_xor(sum, 4);
        sum += __shfl_xor(sum, 8);
        const float inv = 1.0f / sum;
        #pragma unroll
        for (int t = 0; t < 9; ++t) s[t][j] *= inv;
    }

    // ---- spill P to LDS in f16 (C-layout -> row-major), pad cols 144..159 ----
    #pragma unroll
    for (int t = 0; t < 9; ++t)
        #pragma unroll
        for (int j = 0; j < 8; ++j)
            Plds[wave][j + 8 * half][t * 16 + l16] = (_Float16)s[t][j];
    #pragma unroll
    for (int j = 0; j < 8; ++j)
        Plds[wave][j + 8 * half][144 + l16] = (_Float16)0.f;

    // ---- O = P(16x160) @ V_win(160x256), B from V^T (contiguous in k) ----
    #pragma unroll 1
    for (int nt = 0; nt < 16; ++nt) {
        v8f o = zero;
        #pragma unroll
        for (int kt = 0; kt < 160; kt += 32) {
            const _Float16* pr = &Plds[wave][l16][kt + 16 * half];
            v8h a0 = *(const v8h*)pr;
            v8h a1 = *(const v8h*)(pr + 8);
            v16h a;
            #pragma unroll
            for (int e = 0; e < 8; ++e) { a[e] = a0[e]; a[8 + e] = a1[e]; }

            // 16-wide K-chunks are fully in-window or fully P==0 (16-aligned),
            // so clamping the chunk base is safe and loads only finite data.
            int cb = k0 + kt + 16 * half;
            if (cb < 0)          cb = 0;
            if (cb > NTOK - 16)  cb = NTOK - 16;
            v16h b = *(const v16h*)(Vt + (size_t)(nt * 16 + l16) * NTOK + cb);

            o = wmma_f16(a, b, o);
        }
        #pragma unroll
        for (int j = 0; j < 8; ++j)
            out[(size_t)(qStart + j + 8 * half) * DIM + nt * 16 + l16] = o[j];
    }
}

// ---------------------------------------------------------------------------
extern "C" void kernel_launch(void* const* d_in, const int* in_sizes, int n_in,
                              void* d_out, int out_size, void* d_ws, size_t ws_size,
                              hipStream_t stream) {
    const float* x  = (const float*)d_in[0];
    const float* Wq = (const float*)d_in[1];
    const float* bq = (const float*)d_in[2];
    const float* Wk = (const float*)d_in[3];
    const float* bk = (const float*)d_in[4];
    const float* Wv = (const float*)d_in[5];
    const float* bv = (const float*)d_in[6];
    _Float16* ws = (_Float16*)d_ws;              // ~8.8 MB of f16 scratch
    float* out = (float*)d_out;

    // layout prep: x -> f16, W^T -> f16
    prep_kernel<<<(NTOK * DIM) / 256, 256, 0, stream>>>(x, Wq, Wk, Wv, ws);
    // 3 * 256 * 4 = 3072 strip-tiles, 8 waves per 256-thread block
    qkv_project_kernel<<<384, 256, 0, stream>>>(bq, bk, bv, ws);
    // 256 query tiles, 4 waves (one tile each) per 128-thread block
    sw_attention_kernel<<<64, 128, 0, stream>>>(ws, out);
}
